// sLSTM_38611755991167
// MI455X (gfx1250) — matrix operand order repaired and checked by reading, not compiled, not run on hardware
//
#include <hip/hip_runtime.h>

// sLSTM fused scan kernel for gfx1250 (MI455X), wave32 + WMMA f16->f32.
//
//   grid = BATCH/16 workgroups, 256 threads (8 wave32) each.
//   Each WG owns a 16-row batch tile and runs the full 2048-step scan.
//   Wave w computes gate columns [64w, 64w+64) as 4 WMMA N-tiles.
//   Software-pipelined: xacc[q] = b + x_s@W^T (2 WMMA/tile) is computed for
//   step s+1 while the workgroup converges on the barrier; the critical path
//   per step is only h-frag LDS loads -> 4 chained R-WMMAs/tile -> gate store
//   -> split barrier (LDS-only wait) -> gate nonlinearity -> h store -> barrier.
//   Global out stores are issued after the barrier and never synced on.

typedef _Float16 f16;
typedef __attribute__((ext_vector_type(16))) _Float16 v16h;
typedef __attribute__((ext_vector_type(8)))  float    v8f;

#define IN   64
#define HID  128
#define G4   512      // 4*HID
#define SEQL 2048
#define BT   16       // batch tile (WMMA M)
#define HS   136      // hbuf row stride in halfs (272B: 16B-aligned, 4-bank rotation)
#define GS   516      // gates row stride in floats (2064B: 16B-aligned; 8*516%64=32)

// Workgroup split barrier that only waits for LDS (DScnt), keeping in-flight
// global loads/stores off the synchronization path.
__device__ __forceinline__ void wg_barrier_lds_only() {
    asm volatile(
        "s_wait_dscnt 0x0\n"
        "s_barrier_signal -1\n"
        "s_barrier_wait -1"
        ::: "memory");
}

__device__ __forceinline__ float log_sigmoid_f(float x) {
    // -softplus(-x), stabilized: min(x,0) - log1p(exp(-|x|))
    return fminf(x, 0.0f) - __logf(1.0f + __expf(-fabsf(x)));
}
__device__ __forceinline__ float sigmoid_f(float x) {
    return 1.0f / (1.0f + __expf(-x));
}
__device__ __forceinline__ float tanh_f(float x) {
    // saturates correctly for large |x| (exp->inf or ->0)
    return 1.0f - 2.0f / (__expf(2.0f * x) + 1.0f);
}

__global__ __launch_bounds__(256) void slstm_scan_kernel(
    const float* __restrict__ x,   // (B, S, IN)
    const float* __restrict__ W,   // (G4, IN)
    const float* __restrict__ R,   // (HID, G4)
    const float* __restrict__ b,   // (G4,)
    float* __restrict__ out)       // (B, S, HID)
{
    __shared__ f16   hbuf[BT * HS];     // h_{t-1} tile, f16, row-major padded
    __shared__ float gates[BT * GS];    // gate pre-activations, fp32

    const int tid  = threadIdx.x;
    const int lane = tid & 31;
    const int wave = tid >> 5;
    const int b0   = blockIdx.x * BT;

    const int r   = lane & 15;          // A-row / B-col / C-col within tile
    const int kb  = (lane >> 4) * 8;    // A-fragment k-base (0 or 8); also C row-base
    const int khi = (lane >> 4) * 16;   // unused; B-fragments use kb like A

    // ---------------- Preload weight fragments into VGPRs (loop-invariant) ----
    v16h Wf[2][4];   // x-projection B-fragments: Bx[k=i][n=g] = W[g][i]
    v16h Rf[4][4];   // recurrence B-fragments:   Br[k][n]    = R[k][n]
    float bias[4];
    #pragma unroll
    for (int q = 0; q < 4; ++q) {
        const int n0  = (wave * 4 + q) * 16;
        const int col = n0 + r;
        bias[q] = b[col];
        #pragma unroll
        for (int t = 0; t < 2; ++t) {
            const float* wp = &W[(size_t)col * IN + t * 32 + kb];   // 2x8 contiguous (k=kb..+7, kb+16..+23)
            v16h f;
            #pragma unroll
            for (int e = 0; e < 16; ++e) f[e] = (f16)wp[e + (e & 8)];
            Wf[t][q] = f;
        }
        #pragma unroll
        for (int t = 0; t < 4; ++t) {
            const float* rp = &R[(size_t)(t * 32 + kb) * G4 + col];  // stride-G4 gather
            v16h f;
            #pragma unroll
            for (int e = 0; e < 16; ++e) f[e] = (f16)rp[(size_t)(e + (e & 8)) * G4];
            Rf[t][q] = f;
        }
    }

    // ---------------- Init h tile to zero; per-thread state registers --------
    for (int i = tid; i < BT * HS; i += 256) hbuf[i] = (f16)0.0f;

    const int er = tid >> 4;            // elementwise: row 0..15
    const int j0 = (tid & 15) * 8;      // elementwise: 8 contiguous hidden cols
    float cs[8], ns[8], ms[8];
    #pragma unroll
    for (int e = 0; e < 8; ++e) { cs[e] = 0.0f; ns[e] = 0.0f; ms[e] = 0.0f; }

    const float* xrow   = &x[(size_t)(b0 + r) * SEQL * IN];
    float*       outrow = &out[((size_t)(b0 + er) * SEQL) * HID + j0];

    // ---------------- Prologue: xacc = b + x_0 @ W^T -------------------------
    v8f xacc[4];
    {
        const float* xp = xrow;                       // s = 0
        v16h ax[2];
        #pragma unroll
        for (int t = 0; t < 2; ++t) {
            const float* p = xp + t * 32 + kb;
            v16h f;
            #pragma unroll
            for (int e = 0; e < 8; ++e) f[e]     = (f16)p[e];
            #pragma unroll
            for (int e = 0; e < 8; ++e) f[8 + e] = (f16)p[16 + e];
            ax[t] = f;
        }
        #pragma unroll
        for (int q = 0; q < 4; ++q) {
            v8f acc;
            #pragma unroll
            for (int a = 0; a < 8; ++a) acc[a] = bias[q];
            acc = __builtin_amdgcn_wmma_f32_16x16x32_f16(false, ax[0], false, Wf[0][q], (short)0, acc, false, false);
            acc = __builtin_amdgcn_wmma_f32_16x16x32_f16(false, ax[1], false, Wf[1][q], (short)0, acc, false, false);
            xacc[q] = acc;
        }
    }

    __syncthreads();   // hbuf zero-init visible

    for (int s = 0; s < SEQL; ++s) {
        // ---- A-fragments for h_{t-1} from LDS (critical path) ----
        v16h ah[4];
        #pragma unroll
        for (int t = 0; t < 4; ++t) {
            const f16* p = &hbuf[r * HS + t * 32 + kb];
            v16h f;
            #pragma unroll
            for (int e = 0; e < 8; ++e) f[e]     = p[e];
            #pragma unroll
            for (int e = 0; e < 8; ++e) f[8 + e] = p[16 + e];
            ah[t] = f;
        }
        // ---- gates = xacc + h @ R : 4 chained WMMAs per N-tile ----
        #pragma unroll
        for (int q = 0; q < 4; ++q) {
            v8f acc = xacc[q];
            #pragma unroll
            for (int t = 0; t < 4; ++t)
                acc = __builtin_amdgcn_wmma_f32_16x16x32_f16(false, ah[t], false, Rf[t][q], (short)0, acc, false, false);
            // C/D layout: element a -> row (a + kb), col (n0 + r)
            const int n0 = (wave * 4 + q) * 16;
            #pragma unroll
            for (int a = 0; a < 8; ++a)
                gates[(a + kb) * GS + n0 + r] = acc[a];
        }

        // ---- off-critical-path: xacc for step s+1 (overlaps barrier wait) ----
        {
            const int sx = (s + 1 < SEQL) ? (s + 1) : s;
            const float* xp = xrow + (size_t)sx * IN;
            if (s + 2 < SEQL) __builtin_prefetch(xp + IN, 0, 3);   // global_prefetch_b8
            v16h ax[2];
            #pragma unroll
            for (int t = 0; t < 2; ++t) {
                const float* p = xp + t * 32 + kb;
                v16h f;
                #pragma unroll
                for (int e = 0; e < 8; ++e) f[e]     = (f16)p[e];
                #pragma unroll
                for (int e = 0; e < 8; ++e) f[8 + e] = (f16)p[16 + e];
                ax[t] = f;
            }
            #pragma unroll
            for (int q = 0; q < 4; ++q) {
                v8f acc;
                #pragma unroll
                for (int a = 0; a < 8; ++a) acc[a] = bias[q];
                acc = __builtin_amdgcn_wmma_f32_16x16x32_f16(false, ax[0], false, Wf[0][q], (short)0, acc, false, false);
                acc = __builtin_amdgcn_wmma_f32_16x16x32_f16(false, ax[1], false, Wf[1][q], (short)0, acc, false, false);
                xacc[q] = acc;
            }
        }

        wg_barrier_lds_only();   // gates visible; hbuf reads drained

        // ---- elementwise state update: 8 elements per thread ----
        const float* grow = &gates[er * GS + j0];
        float hv[8];
        #pragma unroll
        for (int e = 0; e < 8; ++e) {
            const float gi = grow[e];
            const float gf = grow[HID     + e];
            const float go = grow[2 * HID + e];
            const float gz = grow[3 * HID + e];
            const float lf = log_sigmoid_f(gf);
            const float mn = fmaxf(lf + ms[e], gi);
            const float ip = __expf(gi - mn);
            const float fp = __expf(lf + ms[e] - mn);
            const float cn = fp * cs[e] + ip * tanh_f(gz);
            const float nn = fp * ns[e] + ip;
            const float h  = sigmoid_f(go) * tanh_f(cn / nn);
            cs[e] = cn; ns[e] = nn; ms[e] = mn;
            hbuf[er * HS + j0 + e] = (f16)h;   // feeds next step's A-fragments
            hv[e] = h;
        }

        wg_barrier_lds_only();   // hbuf(h_t) visible; gates free for next step

        // ---- global output store, decoupled from the scan's sync path ----
        float* og = outrow + (size_t)s * HID;
        #pragma unroll
        for (int e = 0; e < 8; ++e) og[e] = hv[e];
    }
}

extern "C" void kernel_launch(void* const* d_in, const int* in_sizes, int n_in,
                              void* d_out, int out_size, void* d_ws, size_t ws_size,
                              hipStream_t stream) {
    const float* x = (const float*)d_in[0];   // (B, S, IN)
    const float* W = (const float*)d_in[1];   // (G4, IN)
    const float* R = (const float*)d_in[2];   // (HID, G4)
    const float* b = (const float*)d_in[3];   // (G4,)
    float* out = (float*)d_out;               // (B, S, HID)

    const int batch  = in_sizes[0] / (SEQL * IN);   // 256
    const int blocks = batch / BT;                  // 16
    slstm_scan_kernel<<<blocks, 256, 0, stream>>>(x, W, R, b, out);
}
